// AngularPenaltySMLoss_32908039422286
// MI455X (gfx1250) — matrix operand order repaired and checked by reading, not compiled, not run on hardware
//
#include <hip/hip_runtime.h>
#include <math.h>

// ---------------------------------------------------------------------------
// CosFace loss for MI455X (gfx1250): f16 WMMA GEMM with f32 accumulation.
// ---------------------------------------------------------------------------

typedef _Float16 v4h  __attribute__((ext_vector_type(4)));
typedef _Float16 v8h  __attribute__((ext_vector_type(8)));
typedef _Float16 v16h __attribute__((ext_vector_type(16)));
typedef float    v8f  __attribute__((ext_vector_type(8)));

#define B_DIM   1024
#define D_DIM   512
#define C_DIM   100000
#define CP_DIM  100352      // 196 * 512, zero-padded class count
#define SCALE_F 30.0f
#define MARGIN_F 0.4f

__device__ __forceinline__ float lane_reduce_sum32(float v) {
#pragma unroll
  for (int m = 16; m > 0; m >>= 1) v += __shfl_xor(v, m, 32);
  return v;
}

// ---------------------------------------------------------------------------
// Kernel 1: normalize x rows, store fp16 TRANSPOSED xhT[D][B]; zero sums[B].
// One wave32 per row. 128 blocks * 8 waves = 1024 rows.
// ---------------------------------------------------------------------------
__global__ __launch_bounds__(256) void xnorm_kernel(const float* __restrict__ x,
                                                    _Float16* __restrict__ xhT,
                                                    float* __restrict__ sums) {
  const int lane = threadIdx.x & 31;
  const int row  = blockIdx.x * 8 + (threadIdx.x >> 5);
  if (blockIdx.x == 0) {
    for (int i = threadIdx.x; i < B_DIM; i += blockDim.x) sums[i] = 0.0f;
  }
  if (row >= B_DIM) return;

  const float* src = x + (size_t)row * D_DIM;
  float4 v[4];
  float s = 0.0f;
#pragma unroll
  for (int i = 0; i < 4; ++i) {
    v[i] = *(const float4*)(src + i * 128 + lane * 4);
    s += v[i].x * v[i].x + v[i].y * v[i].y + v[i].z * v[i].z + v[i].w * v[i].w;
  }
  s = lane_reduce_sum32(s);
  const float inv = 1.0f / fmaxf(sqrtf(s), 1e-12f);
#pragma unroll
  for (int i = 0; i < 4; ++i) {
    const int k = i * 128 + lane * 4;
    xhT[(size_t)(k + 0) * B_DIM + row] = (_Float16)(v[i].x * inv);
    xhT[(size_t)(k + 1) * B_DIM + row] = (_Float16)(v[i].y * inv);
    xhT[(size_t)(k + 2) * B_DIM + row] = (_Float16)(v[i].z * inv);
    xhT[(size_t)(k + 3) * B_DIM + row] = (_Float16)(v[i].w * inv);
  }
}

// ---------------------------------------------------------------------------
// Kernel 2: normalize W rows, store fp16 row-major wh[CP][D]; zero-pad
// rows in [C, CP) so the GEMM needs no load-side bounds checks.
// One wave32 per class row. CP/8 blocks * 8 waves.
// ---------------------------------------------------------------------------
__global__ __launch_bounds__(256) void wnorm_kernel(const float* __restrict__ w,
                                                    _Float16* __restrict__ wh) {
  const int lane = threadIdx.x & 31;
  const int row  = blockIdx.x * 8 + (threadIdx.x >> 5);
  if (row >= CP_DIM) return;

  _Float16* dst = wh + (size_t)row * D_DIM;
  if (row >= C_DIM) {
    const v4h z = {(_Float16)0.0f, (_Float16)0.0f, (_Float16)0.0f, (_Float16)0.0f};
#pragma unroll
    for (int i = 0; i < 4; ++i) *(v4h*)(dst + i * 128 + lane * 4) = z;
    return;
  }

  const float* src = w + (size_t)row * D_DIM;
  float4 v[4];
  float s = 0.0f;
#pragma unroll
  for (int i = 0; i < 4; ++i) {
    v[i] = *(const float4*)(src + i * 128 + lane * 4);
    s += v[i].x * v[i].x + v[i].y * v[i].y + v[i].z * v[i].z + v[i].w * v[i].w;
  }
  s = lane_reduce_sum32(s);
  const float inv = 1.0f / fmaxf(sqrtf(s), 1e-12f);
#pragma unroll
  for (int i = 0; i < 4; ++i) {
    v4h h = {(_Float16)(v[i].x * inv), (_Float16)(v[i].y * inv),
             (_Float16)(v[i].z * inv), (_Float16)(v[i].w * inv)};
    *(v4h*)(dst + i * 128 + lane * 4) = h;
  }
}

// ---------------------------------------------------------------------------
// Kernel 3: WMMA GEMM. D(16x16 f32) = wh_tile(16x32 f16, M=classes)
//                                   x xhT_tile(32x16 f16, N=batch)
// Wave computes 64 classes x 16 batch (4 accumulators, B-frag reused).
// Block = 8 waves -> 512 classes x 16 batch. Grid = (CP/512, B/16).
// Epilogue: LDS transpose -> coalesced stores of cosine + fused exp-sum.
// ---------------------------------------------------------------------------
__global__ __launch_bounds__(256) void cosface_gemm_kernel(
    const _Float16* __restrict__ xhT, const _Float16* __restrict__ wh,
    float* __restrict__ out, float* __restrict__ sums) {
  __shared__ float lds[8][16 * 68];   // per-wave [16 batch][64 cls + 4 pad]

  const int lane = threadIdx.x & 31;
  const int wv   = threadIdx.x >> 5;
  const int b0   = blockIdx.y * 16;
  const int c0   = blockIdx.x * 512 + wv * 64;

  const int mrow  = lane & 15;             // A-frag: lane -> M row (per half)
  const int kbase = (lane < 16) ? 0 : 8;   // A-frag: K sub-offset per lane half

  v8f acc[4] = {};

  for (int kk = 0; kk < 16; ++kk) {
    const int k0 = kk * 32;
    // B fragment (xhT): lane -> N col b0+lane%16; K = kbase+s (s<8), 16+kbase+s-8 (s>=8).
    const _Float16* bp = xhT + (size_t)(k0 + kbase) * B_DIM + b0 + mrow;
    v16h bfrag;
#pragma unroll
    for (int s = 0; s < 8; ++s) {
      bfrag[s]     = bp[(size_t)s * B_DIM];
      bfrag[s + 8] = bp[(size_t)(s + 16) * B_DIM];
    }

#pragma unroll
    for (int t = 0; t < 4; ++t) {
      // A fragment (wh): lane -> row M, two contiguous b128 loads along K.
      const _Float16* ap =
          wh + (size_t)(c0 + t * 16 + mrow) * D_DIM + k0 + kbase;
      const v8h alo = *(const v8h*)(ap);
      const v8h ahi = *(const v8h*)(ap + 16);
      const v16h afrag = __builtin_shufflevector(
          alo, ahi, 0, 1, 2, 3, 4, 5, 6, 7, 8, 9, 10, 11, 12, 13, 14, 15);
      acc[t] = __builtin_amdgcn_wmma_f32_16x16x32_f16(
          false, afrag, false, bfrag, (short)0, acc[t], false, false);
    }
  }

  // ---- Epilogue: stage D tiles (class-major) into LDS batch-major ----
  float* tile = lds[wv];
  const int batch = lane & 15;
  const int radd  = (lane < 16) ? 0 : 8;
#pragma unroll
  for (int t = 0; t < 4; ++t) {
#pragma unroll
    for (int r = 0; r < 8; ++r) {
      tile[batch * 68 + t * 16 + r + radd] = acc[t][r];
    }
  }
  __syncthreads();

  // Coalesced cosine stores: out[1 + b*C + c], consecutive lanes -> consec c.
  float* cosOut = out + 1;
#pragma unroll
  for (int i = 0; i < 32; ++i) {
    const int e  = i * 32 + lane;     // 0..1023 over [16 batch][64 cls]
    const int bb = e >> 6;
    const int cc = e & 63;
    const int c  = c0 + cc;
    if (c < C_DIM) cosOut[(size_t)(b0 + bb) * C_DIM + c] = tile[bb * 68 + cc];
  }

  // Fused partial exp-sum per batch row: avoids re-reading the 410 MB stream.
  {
    const int bb = lane & 15;
    const int h  = lane >> 4;
    float p = 0.0f;
    for (int cc = h * 32; cc < h * 32 + 32; ++cc) {
      if (c0 + cc < C_DIM) p += __expf(SCALE_F * tile[bb * 68 + cc]);
    }
    p += __shfl_xor(p, 16, 32);
    if (lane < 16) atomicAdd(&sums[b0 + bb], p);
  }
}

// ---------------------------------------------------------------------------
// Kernel 4: cosface margin + log-softmax denominator + mean reduction.
// ---------------------------------------------------------------------------
__global__ __launch_bounds__(256) void loss_kernel(const int* __restrict__ label,
                                                   const float* __restrict__ sums,
                                                   float* __restrict__ out) {
  __shared__ float red[256];
  const float* cosOut = out + 1;
  float acc = 0.0f;
  for (int b = threadIdx.x; b < B_DIM; b += 256) {
    const int lbl    = label[b];
    const float tgt  = cosOut[(size_t)b * C_DIM + lbl];
    const float num  = SCALE_F * (tgt - MARGIN_F);
    const float excl = sums[b] - __expf(SCALE_F * tgt);
    const float den  = __expf(num) + excl;
    acc += num - __logf(den);
  }
  red[threadIdx.x] = acc;
  __syncthreads();
  for (int s = 128; s > 0; s >>= 1) {
    if (threadIdx.x < s) red[threadIdx.x] += red[threadIdx.x + s];
    __syncthreads();
  }
  if (threadIdx.x == 0) out[0] = -red[0] / (float)B_DIM;
}

// ---------------------------------------------------------------------------
extern "C" void kernel_launch(void* const* d_in, const int* in_sizes, int n_in,
                              void* d_out, int out_size, void* d_ws,
                              size_t ws_size, hipStream_t stream) {
  (void)in_sizes; (void)n_in; (void)out_size; (void)ws_size;
  const float* x     = (const float*)d_in[0];
  const int*   label = (const int*)d_in[1];
  const float* w     = (const float*)d_in[2];
  float*       out   = (float*)d_out;

  char* ws = (char*)d_ws;
  _Float16* xhT = (_Float16*)(ws);                       // 512*1024*2   = 1.0 MB
  float*    sums = (float*)(ws + (1u << 20));            // 1024*4       = 4 KB
  _Float16* wh  = (_Float16*)(ws + (2u << 20));          // 100352*512*2 = 98 MB

  xnorm_kernel<<<128, 256, 0, stream>>>(x, xhT, sums);
  wnorm_kernel<<<CP_DIM / 8, 256, 0, stream>>>(w, wh);
  dim3 grid(CP_DIM / 512, B_DIM / 16);                   // (196, 64)
  cosface_gemm_kernel<<<grid, 256, 0, stream>>>(xhT, wh, out, sums);
  loss_kernel<<<1, 256, 0, stream>>>(label, sums, out);
}
